// DualGraphSAGEModel_23845658427621
// MI455X (gfx1250) — matrix operand order repaired and checked by reading, not compiled, not run on hardware
//
#include <hip/hip_runtime.h>
#include <hip/hip_bf16.h>

// ---------------------------------------------------------------------------
// Dual GraphSAGE + MLP for MI455X (gfx1250, wave32, WMMA).
//
// mean_neigh(x) @ Wn == mean_neigh(x @ Wn)  =>  per SAGE layer:
//     t_self  = x @ W_self (+ b)                 [WMMA bf16 GEMM]
//     t_neigh = x @ W_neigh                      [WMMA bf16 GEMM]
//     out     = t_self + mean_scatter(t_neigh)   [fp32 L2 atomics + combine]
// Aggregating post-GEMM makes the 128->32 layers scatter 32 dims, not 128.
// GEMM is templated on <NKT, NACC> so the inner loop is a branch-free,
// fully unrolled load/convert/WMMA stream (loads clause + overlap VALU).
// ---------------------------------------------------------------------------

typedef __attribute__((ext_vector_type(16))) __bf16          v16bf;
typedef __attribute__((ext_vector_type(8)))  float           v8f;
typedef __attribute__((ext_vector_type(16))) unsigned short  v16u;

__device__ __forceinline__ unsigned short f32_to_bf16_rne(float f) {
  unsigned int u = __float_as_uint(f);
  u += 0x7FFFu + ((u >> 16) & 1u);          // round-to-nearest-even
  return (unsigned short)(u >> 16);
}

// ---- pack f32 weight W[K][N] (row-major) into WMMA bf16 B fragments -------
// Fragment (kt, nt): 32 lanes x 16 bf16.  Lane L: n = L&15, half = L>>4,
// element e: k = kt*32 + half*16 + e   (B rows striped across lanes).
__global__ void pack_weight_bf16(const float* __restrict__ W,
                                 unsigned short* __restrict__ P,
                                 int K, int N, int nshift) {
  int tid = blockIdx.x * blockDim.x + threadIdx.x;
  if (tid >= K * N) return;
  int k = tid >> nshift;
  int n = tid & (N - 1);
  int kt = k >> 5, kin = k & 31;
  int nt = n >> 4, nin = n & 15;
  int half = kin >> 4, e = kin & 15;
  int lane = half * 16 + nin;
  int nnt = N >> 4;
  size_t idx = (((size_t)(kt * nnt + nt) * 32) + lane) * 16 + e;
  P[idx] = f32_to_bf16_rne(W[tid]);
}

// ---- WMMA GEMM: C[nrows x Nout] = A[nrows x NKT*32](f32->bf16) @ Bp (+b) --
// One wave per 16-row strip; 8 waves (128 rows) per block; NACC n-tiles of
// f32 accumulators per pass.  Fully unrolled, branch-free inner loops.
template <int NKT, int NACC>
__global__ __launch_bounds__(256)
void wmma_gemm(const float* __restrict__ A, int lda,
               const unsigned short* __restrict__ Bp,
               const float* __restrict__ bias,
               float* __restrict__ C, int ldc,
               int nrows, int Nout, int relu) {
  const int lane = threadIdx.x & 31;
  const int wave = threadIdx.x >> 5;
  const int row0 = (blockIdx.x * 8 + wave) * 16;
  if (row0 >= nrows) return;                 // wave-uniform: EXEC stays full
  const int m    = lane & 15;
  const int half = lane >> 4;
  const float* arow = A + (size_t)(row0 + m) * lda;
  const int nnt = Nout >> 4;

  for (int nb = 0; nb < nnt; nb += NACC) {
    v8f acc[NACC];
#pragma unroll
    for (int j = 0; j < NACC; ++j)
#pragma unroll
      for (int r = 0; r < 8; ++r) acc[j][r] = 0.0f;

#pragma unroll
    for (int kt = 0; kt < NKT; ++kt) {
      // stage all B fragments for this k-tile first (loads clause together)
      v16u bu[NACC];
#pragma unroll
      for (int j = 0; j < NACC; ++j)
        bu[j] = *(const v16u*)(Bp + (((size_t)(kt * nnt + nb + j) * 32) + lane) * 16);

      // A fragment (contiguous k order, same as packed B): lane holds row m,
      // k = kt*32 + half*16 + 0..15
      const float* ap = arow + kt * 32 + half * 16;
      v16u au;
#pragma unroll
      for (int e2 = 0; e2 < 8; ++e2) au[e2]     = f32_to_bf16_rne(ap[e2]);
#pragma unroll
      for (int e2 = 0; e2 < 8; ++e2) au[8 + e2] = f32_to_bf16_rne(ap[8 + e2]);
      const v16bf af = __builtin_bit_cast(v16bf, au);

#pragma unroll
      for (int j = 0; j < NACC; ++j)
        acc[j] = __builtin_amdgcn_wmma_f32_16x16x32_bf16(
            false, af, false, __builtin_bit_cast(v16bf, bu[j]),
            (short)0, acc[j], false, false);
    }

    // C/D layout: VGPR r -> row = row0 + half*8 + r, col = ntile*16 + (lane&15)
#pragma unroll
    for (int j = 0; j < NACC; ++j) {
      const int col = (nb + j) * 16 + m;
      const float bv = bias ? bias[col] : 0.0f;
      float* cp = C + (size_t)(row0 + half * 8) * ldc + col;
#pragma unroll
      for (int r = 0; r < 8; ++r) {
        float v = acc[j][r] + bv;
        if (relu) v = fmaxf(v, 0.0f);
        cp[(size_t)r * ldc] = v;
      }
    }
  }
}

// ---- edge / elementwise kernels -------------------------------------------
__global__ void zero_f32(float* __restrict__ p, long n) {
  long t = (long)blockIdx.x * blockDim.x + threadIdx.x;
  if (t < n) p[t] = 0.0f;
}

__global__ void deg_count(const int* __restrict__ dst, float* __restrict__ deg, int nE) {
  int t = blockIdx.x * blockDim.x + threadIdx.x;
  if (t < nE) atomicAdd(&deg[dst[t]], 1.0f);
}

__global__ void deg_recip(float* __restrict__ deg, int n) {
  int t = blockIdx.x * blockDim.x + threadIdx.x;
  if (t < n) deg[t] = 1.0f / fmaxf(deg[t], 1.0f);
}

// one thread = one (edge, 4-dim chunk): float4 gather + 4 fp32 atomics to L2
__global__ void scatter_add4(const float* __restrict__ x, const int* __restrict__ src,
                             const int* __restrict__ dst, float* __restrict__ acc,
                             int d, int dvshift, long total) {
  long t = (long)blockIdx.x * blockDim.x + threadIdx.x;
  if (t >= total) return;
  long e  = t >> dvshift;
  int  c4 = (int)((t & ((1l << dvshift) - 1)) << 2);
  int  s  = src[e], q = dst[e];
  const float4 v = *(const float4*)(x + (size_t)s * d + c4);
  float* ap = acc + (size_t)q * d + c4;
  atomicAdd(ap + 0, v.x);
  atomicAdd(ap + 1, v.y);
  atomicAdd(ap + 2, v.z);
  atomicAdd(ap + 3, v.w);
}

// out[i*ldo + col0 + c] = (relu?)(tself[i*d+c] + accum[i*d+c] * rdeg[i])
__global__ void combine(const float* __restrict__ tself, const float* __restrict__ accum,
                        const float* __restrict__ rdeg, float* __restrict__ out,
                        int ldo, int col0, int dshift, long total, int relu) {
  long t = (long)blockIdx.x * blockDim.x + threadIdx.x;
  if (t >= total) return;
  long i = t >> dshift;
  int  c = (int)(t & ((1l << dshift) - 1));
  float v = tself[t] + accum[t] * rdeg[i];
  if (relu) v = fmaxf(v, 0.0f);
  out[i * (long)ldo + col0 + c] = v;
}

// ---------------------------------------------------------------------------
static inline int cdiv_host(long a, long b) { return (int)((a + b - 1) / b); }

extern "C" void kernel_launch(void* const* d_in, const int* in_sizes, int n_in,
                              void* d_out, int out_size, void* d_ws, size_t ws_size,
                              hipStream_t stream) {
  const float* ori   = (const float*)d_in[0];
  const float* strc  = (const float*)d_in[1];
  const int*   src   = (const int*)d_in[2];
  const int*   dst   = (const int*)d_in[3];
  const float* W1o_s = (const float*)d_in[4];
  const float* W1o_n = (const float*)d_in[5];
  const float* b1o   = (const float*)d_in[6];
  const float* W2o_s = (const float*)d_in[7];
  const float* W2o_n = (const float*)d_in[8];
  const float* b2o   = (const float*)d_in[9];
  const float* W1s_s = (const float*)d_in[10];
  const float* W1s_n = (const float*)d_in[11];
  const float* b1s   = (const float*)d_in[12];
  const float* W2s_s = (const float*)d_in[13];
  const float* W2s_n = (const float*)d_in[14];
  const float* b2s   = (const float*)d_in[15];
  const float* Wm1   = (const float*)d_in[16];
  const float* bm1   = (const float*)d_in[17];
  const float* Wm2   = (const float*)d_in[18];
  const float* bm2   = (const float*)d_in[19];
  float* out = (float*)d_out;

  const int N = in_sizes[0] / 128;   // 100000
  const int E = in_sizes[2];         // 1600000

  // workspace layout (all region sizes multiples of 32B)
  float* bufA = (float*)d_ws;                    // N*128 (t_self / mlp-hidden lo)
  float* bufB = bufA + (size_t)N * 128;          // N*128 (t_neigh / h1 / hidden hi)
  float* bufC = bufB + (size_t)N * 128;          // N*128 (scatter accumulator)
  float* hcat = bufC + (size_t)N * 128;          // N*64  (concat h_ori|h_struc)
  float* rdeg = hcat + (size_t)N * 64;           // N
  unsigned short* pw = (unsigned short*)(rdeg + N);
  unsigned short* p_w1os = pw;                   // 128*128
  unsigned short* p_w1on = p_w1os + 16384;       // 128*128
  unsigned short* p_w2os = p_w1on + 16384;       // 128*32
  unsigned short* p_w2on = p_w2os + 4096;        // 128*32
  unsigned short* p_w1ss = p_w2on + 4096;        // 64*128
  unsigned short* p_w1sn = p_w1ss + 8192;        // 64*128
  unsigned short* p_w2ss = p_w1sn + 8192;        // 128*32
  unsigned short* p_w2sn = p_w2ss + 4096;        // 128*32
  unsigned short* p_wm1  = p_w2sn + 4096;        // 64*256
  unsigned short* p_wm2  = p_wm1  + 16384;       // 256*32

  // ---- pack weights (bf16, WMMA B-fragment layout) ----
  pack_weight_bf16<<<cdiv_host(128 * 128, 256), 256, 0, stream>>>(W1o_s, p_w1os, 128, 128, 7);
  pack_weight_bf16<<<cdiv_host(128 * 128, 256), 256, 0, stream>>>(W1o_n, p_w1on, 128, 128, 7);
  pack_weight_bf16<<<cdiv_host(128 * 32, 256), 256, 0, stream>>>(W2o_s, p_w2os, 128, 32, 5);
  pack_weight_bf16<<<cdiv_host(128 * 32, 256), 256, 0, stream>>>(W2o_n, p_w2on, 128, 32, 5);
  pack_weight_bf16<<<cdiv_host(64 * 128, 256), 256, 0, stream>>>(W1s_s, p_w1ss, 64, 128, 7);
  pack_weight_bf16<<<cdiv_host(64 * 128, 256), 256, 0, stream>>>(W1s_n, p_w1sn, 64, 128, 7);
  pack_weight_bf16<<<cdiv_host(128 * 32, 256), 256, 0, stream>>>(W2s_s, p_w2ss, 128, 32, 5);
  pack_weight_bf16<<<cdiv_host(128 * 32, 256), 256, 0, stream>>>(W2s_n, p_w2sn, 128, 32, 5);
  pack_weight_bf16<<<cdiv_host(64 * 256, 256), 256, 0, stream>>>(Wm1, p_wm1, 64, 256, 8);
  pack_weight_bf16<<<cdiv_host(256 * 32, 256), 256, 0, stream>>>(Wm2, p_wm2, 256, 32, 5);

  // ---- reciprocal in-degree (shared by all 4 aggregations) ----
  zero_f32<<<cdiv_host(N, 256), 256, 0, stream>>>(rdeg, N);
  deg_count<<<cdiv_host(E, 256), 256, 0, stream>>>(dst, rdeg, E);
  deg_recip<<<cdiv_host(N, 256), 256, 0, stream>>>(rdeg, N);

  const int gb = cdiv_host(cdiv_host(N, 16), 8);  // GEMM blocks (8 waves/block)

  // ================= ori branch =================
  // layer 1: 128 -> 128, relu after combine
  wmma_gemm<4, 4><<<gb, 256, 0, stream>>>(ori, 128, p_w1os, b1o, bufA, 128, N, 128, 0);
  wmma_gemm<4, 4><<<gb, 256, 0, stream>>>(ori, 128, p_w1on, nullptr, bufB, 128, N, 128, 0);
  zero_f32<<<cdiv_host((long)N * 128, 256), 256, 0, stream>>>(bufC, (long)N * 128);
  scatter_add4<<<cdiv_host((long)E * 32, 256), 256, 0, stream>>>(bufB, src, dst, bufC, 128, 5, (long)E * 32);
  combine<<<cdiv_host((long)N * 128, 256), 256, 0, stream>>>(bufA, bufC, rdeg, bufB, 128, 0, 7, (long)N * 128, 1);
  // layer 2: 128 -> 32, into hcat[:, 0:32]
  wmma_gemm<4, 2><<<gb, 256, 0, stream>>>(bufB, 128, p_w2os, b2o, bufA, 32, N, 32, 0);
  wmma_gemm<4, 2><<<gb, 256, 0, stream>>>(bufB, 128, p_w2on, nullptr, bufA + (size_t)N * 32, 32, N, 32, 0);
  zero_f32<<<cdiv_host((long)N * 32, 256), 256, 0, stream>>>(bufC, (long)N * 32);
  scatter_add4<<<cdiv_host((long)E * 8, 256), 256, 0, stream>>>(bufA + (size_t)N * 32, src, dst, bufC, 32, 3, (long)E * 8);
  combine<<<cdiv_host((long)N * 32, 256), 256, 0, stream>>>(bufA, bufC, rdeg, hcat, 64, 0, 5, (long)N * 32, 0);

  // ================= struc branch =================
  // layer 1: 64 -> 128, relu after combine
  wmma_gemm<2, 4><<<gb, 256, 0, stream>>>(strc, 64, p_w1ss, b1s, bufA, 128, N, 128, 0);
  wmma_gemm<2, 4><<<gb, 256, 0, stream>>>(strc, 64, p_w1sn, nullptr, bufB, 128, N, 128, 0);
  zero_f32<<<cdiv_host((long)N * 128, 256), 256, 0, stream>>>(bufC, (long)N * 128);
  scatter_add4<<<cdiv_host((long)E * 32, 256), 256, 0, stream>>>(bufB, src, dst, bufC, 128, 5, (long)E * 32);
  combine<<<cdiv_host((long)N * 128, 256), 256, 0, stream>>>(bufA, bufC, rdeg, bufB, 128, 0, 7, (long)N * 128, 1);
  // layer 2: 128 -> 32, into hcat[:, 32:64]
  wmma_gemm<4, 2><<<gb, 256, 0, stream>>>(bufB, 128, p_w2ss, b2s, bufA, 32, N, 32, 0);
  wmma_gemm<4, 2><<<gb, 256, 0, stream>>>(bufB, 128, p_w2sn, nullptr, bufA + (size_t)N * 32, 32, N, 32, 0);
  zero_f32<<<cdiv_host((long)N * 32, 256), 256, 0, stream>>>(bufC, (long)N * 32);
  scatter_add4<<<cdiv_host((long)E * 8, 256), 256, 0, stream>>>(bufA + (size_t)N * 32, src, dst, bufC, 32, 3, (long)E * 8);
  combine<<<cdiv_host((long)N * 32, 256), 256, 0, stream>>>(bufA, bufC, rdeg, hcat, 64, 32, 5, (long)N * 32, 0);

  // ================= MLP =================
  float* hm = bufA;  // N*256 spans bufA..bufB (contiguous), both free now
  wmma_gemm<2, 8><<<gb, 256, 0, stream>>>(hcat, 64, p_wm1, bm1, hm, 256, N, 256, 1);
  wmma_gemm<8, 2><<<gb, 256, 0, stream>>>(hm, 256, p_wm2, bm2, out, 32, N, 32, 0);
}
